// EdgeAugmentedGraphTransformer_60241211294241
// MI455X (gfx1250) — matrix operand order repaired and checked by reading, not compiled, not run on hardware
//
#include <hip/hip_runtime.h>
#include <hip/hip_bf16.h>

#define HID 32
#define HEADS 8
#define NLAYERS 4
#define BB 64
#define NNODE 128
#define NV (BB*NNODE)        // 8192 node rows
#define NE2 (NNODE*NNODE)    // 16384
#define NE (BB*NE2)          // 1048576 edge rows

typedef __attribute__((ext_vector_type(16))) _Float16 v16h;
typedef __attribute__((ext_vector_type(8)))  float    v8f;

__device__ __forceinline__ float fsigmoid(float x){ return 1.0f/(1.0f+__expf(-x)); }
__device__ __forceinline__ float felu(float x){ return x>0.0f? x : (__expf(x)-1.0f); }
__device__ __forceinline__ float frelu(float x){ return x>0.0f? x:0.0f; }

__device__ __forceinline__ void ln32(const float* x, const float* g, const float* b, float* y){
  float s=0.f,s2=0.f;
  #pragma unroll
  for(int i=0;i<32;i++){ s+=x[i]; s2+=x[i]*x[i]; }
  float mu=s*0.03125f;
  float var=s2*0.03125f-mu*mu;
  float inv=rsqrtf(var+1e-5f);
  #pragma unroll
  for(int i=0;i<32;i++) y[i]=(x[i]-mu)*inv*g[i]+b[i];
}

// WMMA fragment helpers (CDNA5 16x16x32 f16 layouts)
__device__ __forceinline__ v16h loadA16(const _Float16* row, int k0){
  v16h a;
  ((float4*)&a)[0] = *(const float4*)(row+k0);     // k = k0..k0+7   -> elems 0..7
  ((float4*)&a)[1] = *(const float4*)(row+k0+8);   // k = k0+8..+15  -> elems 8..15
  return a;
}
__device__ __forceinline__ v16h loadB16(const _Float16* p){
  v16h a;
  ((float4*)&a)[0] = *(const float4*)(p);
  ((float4*)&a)[1] = *(const float4*)(p+8);
  return a;
}
#define WMMA_F16(a,b,c) __builtin_amdgcn_wmma_f32_16x16x32_f16(false,(a),false,(b),(short)0,(c),false,false)

// ---------------- input embeddings ----------------

__global__ void k_pdf_emb(const float* __restrict__ pdf, const float* __restrict__ w,
                          const float* __restrict__ b, float* __restrict__ out){
  __shared__ float red[128];
  int bidx = blockIdx.x; int tid = threadIdx.x;
  for(int o=0;o<10;o++){
    float s=0.f;
    for(int k=tid;k<3000;k+=128) s += pdf[bidx*3000+k]*w[k*10+o];
    red[tid]=s; __syncthreads();
    for(int off=64;off>0;off>>=1){ if(tid<off) red[tid]+=red[tid+off]; __syncthreads(); }
    if(tid==0) out[bidx*10+o]=red[0]+b[o];
    __syncthreads();
  }
}

__global__ void k_node_emb(const float* __restrict__ species, const float* __restrict__ pdfe,
    const float* __restrict__ w1,const float* __restrict__ b1,
    const float* __restrict__ w2,const float* __restrict__ b2,
    float* __restrict__ hbuf){
  int r = blockIdx.x*blockDim.x + threadIdx.x; if(r>=NV) return;
  int bb = r/NNODE;
  float x[11]; x[0]=species[r];
  #pragma unroll
  for(int i=0;i<10;i++) x[1+i]=pdfe[bb*10+i];
  float h1[16];
  for(int o=0;o<16;o++){ float s=b1[o]; for(int k=0;k<11;k++) s+=x[k]*w1[k*16+o]; h1[o]=frelu(s); }
  for(int o=0;o<32;o++){ float s=b2[o]; for(int k=0;k<16;k++) s+=h1[k]*w2[k*32+o]; hbuf[(size_t)r*32+o]=s; }
}

// ---- edge input MLP [11->16 relu ->32], WMMA-tiled: 16 rows per wave ----

__global__ void __launch_bounds__(128) k_edge_emb_w(
  const float* __restrict__ adj, const float* __restrict__ pdfe,
  const float* w1,const float* b1,const float* w2,const float* b2,
  float* __restrict__ ebuf){
  __shared__ __align__(16) _Float16 sW1[16*32]; // [n][k], k<11 valid
  __shared__ __align__(16) _Float16 sW2[32*32]; // [n][k], k<16 valid
  __shared__ float sB1[16], sB2[32];
  __shared__ __align__(16) _Float16 sA[4][16*32];
  __shared__ float sD[4][16*32];
  int tid=threadIdx.x;
  int wave=tid>>5, lane=tid&31, m=lane&15, hf=lane>>4;
  size_t R=(size_t)blockIdx.x*64 + (size_t)wave*16;
  size_t e=R+m;
  // early global loads (overlap with LDS weight preload)
  float x0=0.f, pe[10];
  if(hf==0){
    x0=adj[e];
    int bbatch=(int)(e/NE2);
    #pragma unroll
    for(int i2=0;i2<10;i2++) pe[i2]=pdfe[bbatch*10+i2];
  }
  for(int idx=tid; idx<16*32; idx+=128){ int n=idx>>5,k=idx&31; sW1[idx]=(_Float16)(k<11? w1[k*16+n]:0.0f); }
  for(int idx=tid; idx<32*32; idx+=128){ int n=idx>>5,k=idx&31; sW2[idx]=(_Float16)(k<16? w2[k*32+n]:0.0f); }
  if(tid<16) sB1[tid]=b1[tid];
  if(tid<32) sB2[tid]=b2[tid];
  _Float16* myA=sA[wave]; float* myD=sD[wave];
  // stage A [16][32]: k<11 = x, else 0
  if(hf==0){
    myA[m*32+0]=(_Float16)x0;
    #pragma unroll
    for(int i2=0;i2<10;i2++) myA[m*32+1+i2]=(_Float16)pe[i2];
    #pragma unroll
    for(int k=11;k<16;k++) myA[m*32+k]=(_Float16)0.0f;
  } else {
    #pragma unroll
    for(int k=16;k<32;k++) myA[m*32+k]=(_Float16)0.0f;
  }
  __syncthreads();
  // GEMM1 -> 16x16
  v16h a1=loadA16(&myA[m*32], hf*16);
  v16h bA=loadB16(&sW1[m*32 + hf*16]);
  v8f c={}; c=WMMA_F16(a1,bA,c);
  #pragma unroll
  for(int i2=0;i2<8;i2++) myD[(hf*8+i2)*16 + m] = c[i2];
  __syncthreads();
  // relu + restage [16][32] f16 (k<16 valid)
  #pragma unroll
  for(int t=0;t<8;t++){
    float v=myD[m*16 + hf*8 + t] + sB1[hf*8+t];
    myA[m*32 + hf*8 + t] = (_Float16)frelu(v);
    myA[m*32 + 16 + hf*8 + t] = (_Float16)0.0f;
  }
  __syncthreads();
  // GEMM2 -> 16x32
  v16h a2=loadA16(&myA[m*32], hf*16);
  v16h b20=loadB16(&sW2[m*32 + hf*16]);
  v16h b21=loadB16(&sW2[(m+16)*32 + hf*16]);
  v8f d0={},d1={};
  d0=WMMA_F16(a2,b20,d0); d1=WMMA_F16(a2,b21,d1);
  #pragma unroll
  for(int i2=0;i2<8;i2++){ myD[(hf*8+i2)*32+m]=d0[i2]; myD[(hf*8+i2)*32+m+16]=d1[i2]; }
  __syncthreads();
  {
    float* out=&ebuf[e*32 + hf*16];
    #pragma unroll
    for(int t=0;t<16;t++) out[t]=myD[m*32+hf*16+t]+sB2[hf*16+t];
  }
}

// ---------------- per-layer node pre (LN + QKV) ----------------

__global__ void k_qkv(const float* __restrict__ hbuf,
   const float* lng,const float* lnb,
   const float* wq,const float* bq,const float* wk,const float* bk,
   const float* wv,const float* bv,
   float* __restrict__ qbuf,float* __restrict__ kbuf,float* __restrict__ vbuf){
  int r = blockIdx.x*blockDim.x+threadIdx.x; if(r>=NV) return;
  float x[32], xn[32];
  #pragma unroll
  for(int i=0;i<32;i++) x[i]=hbuf[(size_t)r*32+i];
  ln32(x,lng,lnb,xn);
  for(int o=0;o<32;o++){
    float sq=bq[o],sk=bk[o],sv=bv[o];
    for(int k=0;k<32;k++){ float xv=xn[k]; sq+=xv*wq[k*32+o]; sk+=xv*wk[k*32+o]; sv+=xv*wv[k*32+o]; }
    qbuf[(size_t)r*32+o]=sq; kbuf[(size_t)r*32+o]=sk; vbuf[(size_t)r*32+o]=sv;
  }
}

// ---- edge LN + (g|e) projection via one WMMA + qk scores ----
// B tile packs Wg into cols 0..7 and We into cols 8..15.

__global__ void __launch_bounds__(128) k_edge_hg_w(
  const float* __restrict__ ebuf,
  const float* lng,const float* lnb,
  const float* wg,const float* bg,const float* we,const float* be,
  const float* __restrict__ qbuf,const float* __restrict__ kbuf,
  float* __restrict__ Hbuf,float* __restrict__ Gbuf){
  __shared__ __align__(16) _Float16 sWge[16*32]; // [n][k]
  __shared__ float sLg[32], sLb[32], sBg[8], sBe[8];
  __shared__ __align__(16) _Float16 sA[4][16*32];
  __shared__ float sD[4][16*16];
  int tid=threadIdx.x;
  int wave=tid>>5, lane=tid&31, m=lane&15, hf=lane>>4;
  size_t R=(size_t)blockIdx.x*64 + (size_t)wave*16;
  // early residual-row load
  float v[16];
  {
    const float* er=&ebuf[(R+m)*32 + hf*16];
    *(float4*)&v[0]=*(const float4*)(er);
    *(float4*)&v[4]=*(const float4*)(er+4);
    *(float4*)&v[8]=*(const float4*)(er+8);
    *(float4*)&v[12]=*(const float4*)(er+12);
  }
  for(int idx=tid; idx<16*32; idx+=128){
    int n=idx>>5,k=idx&31;
    sWge[idx]=(_Float16)(n<8? wg[k*8+n] : we[k*8+(n-8)]);
  }
  if(tid<32){ sLg[tid]=lng[tid]; sLb[tid]=lnb[tid]; }
  if(tid<8){ sBg[tid]=bg[tid]; sBe[tid]=be[tid]; }
  _Float16* myA=sA[wave]; float* myD=sD[wave];
  // LN(ln_e) -> stage f16 A [16][32]
  {
    float s1=0.f,s2=0.f;
    #pragma unroll
    for(int t=0;t<16;t++){ s1+=v[t]; s2+=v[t]*v[t]; }
    s1+=__shfl_xor(s1,16); s2+=__shfl_xor(s2,16);
    float mu=s1*0.03125f, var=s2*0.03125f-mu*mu, inv=rsqrtf(var+1e-5f);
    #pragma unroll
    for(int t=0;t<16;t++) myA[m*32+hf*16+t]=(_Float16)((v[t]-mu)*inv*sLg[hf*16+t]+sLb[hf*16+t]);
  }
  __syncthreads();
  v16h a=loadA16(&myA[m*32], hf*16);
  v16h bf=loadB16(&sWge[m*32 + hf*16]);
  v8f c={}; c=WMMA_F16(a,bf,c);
  #pragma unroll
  for(int i2=0;i2<8;i2++) myD[(hf*8+i2)*16+m]=c[i2];
  __syncthreads();
  {
    size_t e=R+m;
    int bbatch=(int)(e/NE2); int ij=(int)(e%NE2); int ii=ij/NNODE, jj=ij%NNODE;
    const float* q=&qbuf[((size_t)bbatch*NNODE+ii)*32];
    const float* kk=&kbuf[((size_t)bbatch*NNODE+jj)*32];
    #pragma unroll
    for(int t=0;t<4;t++){
      int h=hf*4+t;
      float qk=0.f;
      #pragma unroll
      for(int d=0;d<4;d++) qk+=q[d*8+h]*kk[d*8+h];
      float Hc=fminf(fmaxf(qk*0.5f,-5.f),5.f) + myD[m*16+8+h] + sBe[h]; // clip(qk/2)+e
      Hbuf[e*8+h]=Hc;
      Gbuf[e*8+h]=fsigmoid(myD[m*16+h]+sBg[h]);
    }
  }
}

// ---------------- softmax over j, fold sigmoid gate into A (stored in Gbuf) ----

__global__ void k_softmax(const float* __restrict__ Hbuf, float* __restrict__ Gbuf){
  int t=blockIdx.x*blockDim.x+threadIdx.x; if(t>=NV*HEADS) return;
  int h=t&7; int r=t>>3;
  size_t base=((size_t)r*NNODE)*8 + h;
  float mx=-1e30f;
  for(int j=0;j<NNODE;j++) mx=fmaxf(mx,Hbuf[base+(size_t)j*8]);
  float s=0.f;
  for(int j=0;j<NNODE;j++) s+=__expf(Hbuf[base+(size_t)j*8]-mx);
  float inv=1.0f/s;
  for(int j=0;j<NNODE;j++){ size_t idx=base+(size_t)j*8; Gbuf[idx]=Gbuf[idx]*__expf(Hbuf[idx]-mx)*inv; }
}

// ---------------- node: attn, lin_n residual, ffn_n, final LN ----------------

__global__ void k_node_upd(float* __restrict__ hbuf,
   const float* __restrict__ Abuf, const float* __restrict__ vbuf,
   const float* wn,const float* bn,
   const float* flng,const float* flnb,const float* fw1,const float* fb1,
   const float* fw2,const float* fb2,
   const float* lfg,const float* lfb){
  int r=blockIdx.x*blockDim.x+threadIdx.x; if(r>=NV) return;
  int bb=r/NNODE;
  float attn[32];
  #pragma unroll
  for(int c=0;c<32;c++) attn[c]=0.f;
  for(int j=0;j<NNODE;j++){
    const float* A=&Abuf[((size_t)r*NNODE+j)*8];
    const float* v=&vbuf[((size_t)bb*NNODE+j)*32];
    #pragma unroll
    for(int c=0;c<32;c++) attn[c]+=A[c&7]*v[c];
  }
  float x[32];
  for(int c=0;c<32;c++){
    float s=bn[c];
    for(int k=0;k<32;k++) s+=attn[k]*wn[k*32+c];
    x[c]=s+hbuf[(size_t)r*32+c];
  }
  float xn[32]; ln32(x,flng,flnb,xn);
  float hid[64];
  for(int o=0;o<64;o++){ float s=fb1[o]; for(int k=0;k<32;k++) s+=xn[k]*fw1[k*64+o]; hid[o]=felu(s); }
  float y[32];
  for(int c=0;c<32;c++){ float s=fb2[c]; for(int k=0;k<64;k++) s+=hid[k]*fw2[k*32+c]; y[c]=x[c]+s; }
  float yn[32]; ln32(y,lfg,lfb,yn);
  #pragma unroll
  for(int c=0;c<32;c++) hbuf[(size_t)r*32+c]=yn[c];
}

// ---------------- WMMA fused edge update + symmetrize-on-write ----------------
// Only upper-triangle rows (j>i) are real work: result written to (i,j) AND (j,i);
// diagonal rows write zeros; strict-lower rows produce nothing (their output is the
// mirror of an upper row). Blocks fully inside the strict lower triangle exit early
// (uniform condition, before any barrier).

__global__ void __launch_bounds__(128) k_edge_upd(
   float* __restrict__ ebuf, const float* __restrict__ Hbuf,
   const float* wle,const float* ble,
   const float* flng,const float* flnb,
   const float* fw1,const float* fb1,const float* fw2,const float* fb2,
   const float* lfg,const float* lfb){
  size_t Rblk=(size_t)blockIdx.x*64;            // 64 rows: fixed (b,i), j in [Jb, Jb+63]
  int ijb=(int)(Rblk%NE2); int ib=ijb/NNODE; int Jb=ijb%NNODE;
  if(ib > Jb+63) return;                        // whole block strictly lower -> dead work

  __shared__ __align__(16) _Float16 sWle[32*32];  // [n][k], Kpad=32 (k<8 valid)
  __shared__ __align__(16) _Float16 sW1[64*32];   // [n][k], K=32
  __shared__ __align__(16) _Float16 sW2[32*64];   // [n][k], K=64
  __shared__ float sBle[32],sB1[64],sB2[32],sG1[32],sBt1[32],sG2[32],sBt2[32];
  __shared__ __align__(16) _Float16 sA[4][16*64]; // per-wave A staging
  __shared__ float sD[4][16*64];                  // per-wave D staging
  int tid=threadIdx.x;
  int wave=tid>>5, lane=tid&31, m=lane&15, hf=lane>>4;
  size_t R=Rblk + (size_t)wave*16;
  size_t e=R+m;
  int jj=Jb + wave*16 + m;                       // column j of this row
  bool up = (jj>ib), diag = (jj==ib);

  // ---- early global loads: residual edge row + H row (upper rows only)
  float erv[16];
  if(up){
    const float* er=&ebuf[e*32 + hf*16];
    *(float4*)&erv[0]=*(const float4*)(er);
    *(float4*)&erv[4]=*(const float4*)(er+4);
    *(float4*)&erv[8]=*(const float4*)(er+8);
    *(float4*)&erv[12]=*(const float4*)(er+12);
  } else {
    #pragma unroll
    for(int t=0;t<16;t++) erv[t]=0.f;
  }
  v16h aH={};
  if(hf==0 && up){
    const float* hp=&Hbuf[e*8];
    float4 h0=*(const float4*)hp, h1=*(const float4*)(hp+4);
    aH[0]=(_Float16)h0.x; aH[1]=(_Float16)h0.y; aH[2]=(_Float16)h0.z; aH[3]=(_Float16)h0.w;
    aH[4]=(_Float16)h1.x; aH[5]=(_Float16)h1.y; aH[6]=(_Float16)h1.z; aH[7]=(_Float16)h1.w;
  }
  // ---- weight preload to LDS (f16, transposed to [n][k])
  for(int idx=tid; idx<32*32; idx+=128){ int n=idx>>5,k=idx&31; sWle[idx]=(_Float16)(k<8? wle[k*32+n]:0.0f); }
  for(int idx=tid; idx<64*32; idx+=128){ int n=idx>>5,k=idx&31; sW1[idx]=(_Float16)fw1[k*64+n]; }
  for(int idx=tid; idx<32*64; idx+=128){ int n=idx>>6,k=idx&63; sW2[idx]=(_Float16)fw2[k*32+n]; }
  if(tid<32){ sBle[tid]=ble[tid]; sB2[tid]=fb2[tid]; sG1[tid]=flng[tid]; sBt1[tid]=flnb[tid]; sG2[tid]=lfg[tid]; sBt2[tid]=lfb[tid]; }
  if(tid<64){ sB1[tid]=fb1[tid]; }
  __syncthreads();

  _Float16* myA=sA[wave]; float* myD=sD[wave];

  // ---- GEMM1: H[16x8 pad 32] @ Wle^T -> 16x32
  v8f c0={},c1={};
  v16h b0=loadB16(&sWle[m*32 + hf*16]);
  v16h b1f=loadB16(&sWle[(m+16)*32 + hf*16]);
  c0=WMMA_F16(aH,b0,c0);
  c1=WMMA_F16(aH,b1f,c1);
  #pragma unroll
  for(int i=0;i<8;i++){ myD[(hf*8+i)*32 + m]=c0[i]; myD[(hf*8+i)*32 + m + 16]=c1[i]; }
  __syncthreads();

  // ---- row phase 1: +bias +edge residual, LN(ffn_e.ln), stage f16 A [16][32]
  float eh[16];
  {
    float s1=0.f,s2=0.f;
    #pragma unroll
    for(int t=0;t<16;t++){ float v=myD[m*32+hf*16+t]+sBle[hf*16+t]+erv[t]; eh[t]=v; s1+=v; s2+=v*v; }
    s1+=__shfl_xor(s1,16); s2+=__shfl_xor(s2,16);
    float mu=s1*0.03125f, var=s2*0.03125f-mu*mu, inv=rsqrtf(var+1e-5f);
    #pragma unroll
    for(int t=0;t<16;t++){ float xn=(eh[t]-mu)*inv*sG1[hf*16+t]+sBt1[hf*16+t]; myA[m*32+hf*16+t]=(_Float16)xn; }
  }
  __syncthreads();

  // ---- GEMM2: [16x32] @ W1 -> 16x64
  v8f d0={},d1={},d2={},d3={};
  {
    v16h a2=loadA16(&myA[m*32], hf*16);
    v16h bb0=loadB16(&sW1[(m   )*32 + hf*16]);
    v16h bb1=loadB16(&sW1[(m+16)*32 + hf*16]);
    v16h bb2=loadB16(&sW1[(m+32)*32 + hf*16]);
    v16h bb3=loadB16(&sW1[(m+48)*32 + hf*16]);
    d0=WMMA_F16(a2,bb0,d0); d1=WMMA_F16(a2,bb1,d1);
    d2=WMMA_F16(a2,bb2,d2); d3=WMMA_F16(a2,bb3,d3);
  }
  __syncthreads();
  #pragma unroll
  for(int i=0;i<8;i++){
    myD[(hf*8+i)*64 + m]      = d0[i];
    myD[(hf*8+i)*64 + m + 16] = d1[i];
    myD[(hf*8+i)*64 + m + 32] = d2[i];
    myD[(hf*8+i)*64 + m + 48] = d3[i];
  }
  __syncthreads();

  // ---- row phase 2: +bias, ELU, stage f16 A [16][64]
  #pragma unroll
  for(int t=0;t<32;t++){
    float v=myD[m*64 + hf*32 + t] + sB1[hf*32+t];
    myA[m*64 + hf*32 + t]=(_Float16)felu(v);
  }
  __syncthreads();

  // ---- GEMM3: [16x64] @ W2 -> 16x32 (K split in two 32-chunks)
  v8f g0={},g1={};
  #pragma unroll
  for(int kc=0;kc<2;kc++){
    v16h a3=loadA16(&myA[m*64], kc*32 + hf*16);
    v16h cb0=loadB16(&sW2[(m   )*64 + kc*32 + hf*16]);
    v16h cb1=loadB16(&sW2[(m+16)*64 + kc*32 + hf*16]);
    g0=WMMA_F16(a3,cb0,g0);
    g1=WMMA_F16(a3,cb1,g1);
  }
  __syncthreads();
  #pragma unroll
  for(int i=0;i<8;i++){ myD[(hf*8+i)*32+m]=g0[i]; myD[(hf*8+i)*32+m+16]=g1[i]; }
  __syncthreads();

  // ---- row phase 3: +bias +e_hat residual, final LN, symmetrized store
  {
    float s1=0.f,s2=0.f; float y[16];
    #pragma unroll
    for(int t=0;t<16;t++){ float v=myD[m*32+hf*16+t]+sB2[hf*16+t]+eh[t]; y[t]=v; s1+=v; s2+=v*v; }
    s1+=__shfl_xor(s1,16); s2+=__shfl_xor(s2,16);
    float mu=s1*0.03125f, var=s2*0.03125f-mu*mu, inv=rsqrtf(var+1e-5f);
    if(up){
      float res[16];
      #pragma unroll
      for(int t=0;t<16;t++) res[t]=(y[t]-mu)*inv*sG2[hf*16+t]+sBt2[hf*16+t];
      float* out=&ebuf[e*32 + hf*16];
      #pragma unroll
      for(int t=0;t<16;t++) out[t]=res[t];
      size_t emir=(e/NE2)*NE2 + (size_t)jj*NNODE + ib;   // (b, j, i)
      float* out2=&ebuf[emir*32 + hf*16];
      #pragma unroll
      for(int t=0;t<16;t++) out2[t]=res[t];
    } else if(diag){
      float* out=&ebuf[e*32 + hf*16];
      #pragma unroll
      for(int t=0;t<16;t++) out[t]=0.f;
    }
  }
}

// ---------------- output heads ----------------

__global__ void k_node_out(const float* __restrict__ hbuf,
  const float* w1,const float* b1,const float* w2,const float* b2,
  float* __restrict__ out){
  int r=blockIdx.x*blockDim.x+threadIdx.x; if(r>=NV) return;
  const float* x=&hbuf[(size_t)r*32];
  float acc=b2[0];
  for(int o=0;o<16;o++){ float s=b1[o]; for(int k=0;k<32;k++) s+=x[k]*w1[k*16+o]; acc+=frelu(s)*w2[o]; }
  out[r]=acc;
}

__global__ void k_edge_out(const float* __restrict__ ebuf,
  const float* w1,const float* b1,const float* w2,const float* b2,
  float* __restrict__ out){
  size_t e=(size_t)blockIdx.x*blockDim.x+threadIdx.x; if(e>=NE) return;
  const float* x=&ebuf[e*32];
  float acc=b2[0];
  for(int o=0;o<16;o++){ float s=b1[o]; for(int k=0;k<32;k++) s+=x[k]*w1[k*16+o]; acc+=frelu(s)*w2[o]; }
  out[e]=acc;
}

// ---------------- host launcher ----------------
// d_in order (recursive dict insertion order):
// 0 atom_species, 1 adj_matrix, 2 pad_mask(unused), 3 pdf, 4 t(unused),
// 5 pdf_emb.w, 6 pdf_emb.b, 7-10 emb_in, 11-14 edge_in, 15-18 emb_out, 19-22 edge_out,
// then per layer L (base p=23+34L):
//  +0,1 ln_e  +2,3 ln_n  +4,5 q  +6,7 k  +8,9 v  +10,11 g  +12,13 e
//  +14,15 lin_n  +16,17 lin_e  +18..23 ffn_e(ln,l1,l2)  +24..29 ffn_n  +30,31 ln_e_f  +32,33 ln_n_f

extern "C" void kernel_launch(void* const* d_in, const int* in_sizes, int n_in,
                              void* d_out, int out_size, void* d_ws, size_t ws_size,
                              hipStream_t stream){
  (void)in_sizes; (void)n_in; (void)out_size; (void)ws_size;
  auto F=[&](int i)->const float*{ return (const float*)d_in[i]; };
  const float* species=F(0);
  const float* adj=F(1);
  const float* pdf=F(3);

  float* ws=(float*)d_ws;
  float* pdfe=ws;                       // 640 (pad to 1024)
  float* hbuf=pdfe+1024;                // NV*32
  float* qbuf=hbuf+(size_t)NV*32;
  float* kbuf=qbuf+(size_t)NV*32;
  float* vbuf=kbuf+(size_t)NV*32;
  float* ebuf=vbuf+(size_t)NV*32;       // NE*32 (128 MB)
  float* Hbuf=ebuf+(size_t)NE*32;       // NE*8  (32 MB)
  float* Gbuf=Hbuf+(size_t)NE*8;        // NE*8  (32 MB)

  k_pdf_emb<<<64,128,0,stream>>>(pdf,F(5),F(6),pdfe);
  k_node_emb<<<NV/128,128,0,stream>>>(species,pdfe,F(7),F(8),F(9),F(10),hbuf);
  k_edge_emb_w<<<NE/64,128,0,stream>>>(adj,pdfe,F(11),F(12),F(13),F(14),ebuf);

  for(int L=0;L<NLAYERS;L++){
    int p=23+34*L;
    k_qkv<<<NV/128,128,0,stream>>>(hbuf,F(p+2),F(p+3),F(p+4),F(p+5),F(p+6),F(p+7),F(p+8),F(p+9),
                                   qbuf,kbuf,vbuf);
    k_edge_hg_w<<<NE/64,128,0,stream>>>(ebuf,F(p+0),F(p+1),F(p+10),F(p+11),F(p+12),F(p+13),
                                        qbuf,kbuf,Hbuf,Gbuf);
    k_softmax<<<(NV*HEADS)/256,256,0,stream>>>(Hbuf,Gbuf);
    k_node_upd<<<NV/128,128,0,stream>>>(hbuf,Gbuf,vbuf,F(p+14),F(p+15),
                                        F(p+24),F(p+25),F(p+26),F(p+27),F(p+28),F(p+29),
                                        F(p+32),F(p+33));
    k_edge_upd<<<NE/64,128,0,stream>>>(ebuf,Hbuf,F(p+16),F(p+17),
                                       F(p+18),F(p+19),F(p+20),F(p+21),F(p+22),F(p+23),
                                       F(p+30),F(p+31));
  }

  k_node_out<<<NV/128,128,0,stream>>>(hbuf,F(15),F(16),F(17),F(18),(float*)d_out);
  k_edge_out<<<NE/256,256,0,stream>>>(ebuf,F(19),F(20),F(21),F(22),(float*)d_out+NV);
}
